// TensorProductLigConvLayer_23854248362256
// MI455X (gfx1250) — matrix operand order repaired and checked by reading, not compiled, not run on hardware
//
#include <hip/hip_runtime.h>
#include <stdint.h>

typedef __attribute__((ext_vector_type(16))) __bf16 v16bf;
typedef __attribute__((ext_vector_type(8)))  __bf16 v8bf;
typedef __attribute__((ext_vector_type(8)))  float  v8f;

#define WMMA_BF16 __builtin_amdgcn_wmma_f32_16x16x32_bf16

constexpr int N_NODES = 4096;
constexpr int E_EDGES = 32768;
// DIM=112, WNUM=6400, 400 column-chunks of 16, 100 stages of 4 chunks.

// ---- CDNA5 async load-to-LDS path (ASYNCcnt) -------------------------------
#if __has_builtin(__builtin_amdgcn_global_load_async_to_lds_b128)
#define HAVE_ASYNC_LDS 1
typedef __attribute__((__vector_size__(4 * sizeof(int)))) int v4i_t;
typedef __attribute__((address_space(1))) v4i_t g_v4i;
typedef __attribute__((address_space(3))) v4i_t l_v4i;
__device__ inline void async_cp16(const void* g, void* l) {
    __builtin_amdgcn_global_load_async_to_lds_b128((g_v4i*)g, (l_v4i*)l, 0, 0);
}
#else
#define HAVE_ASYNC_LDS 0
#endif

__device__ inline void wait_async0() {
#if __has_builtin(__builtin_amdgcn_s_wait_asynccnt)
    __builtin_amdgcn_s_wait_asynccnt(0);
#else
    asm volatile("s_wait_asynccnt 0x0" ::: "memory");
#endif
}

__device__ inline uint32_t packbf2(float a, float b) {
    unsigned short ua = __builtin_bit_cast(unsigned short, (__bf16)a);
    unsigned short ub = __builtin_bit_cast(unsigned short, (__bf16)b);
    return (uint32_t)ua | ((uint32_t)ub << 16);
}

union BFrag { v16bf v; v8bf h[2]; };
__device__ inline v16bf load2x8(const __bf16* p0, const __bf16* p1) {
    BFrag f; f.h[0] = *(const v8bf*)p0; f.h[1] = *(const v8bf*)p1; return f.v;
}

// ---------------- prep: fp32 -> bf16 conversions + zero accumulators --------
__global__ void prep_kernel(const float* __restrict__ W1, const float* __restrict__ W2,
                            const float* __restrict__ EA,
                            __bf16* __restrict__ W1h, __bf16* __restrict__ W2Th,
                            __bf16* __restrict__ EAh,
                            float* __restrict__ sums, float* __restrict__ cnt) {
    const size_t tid = (size_t)blockIdx.x * blockDim.x + threadIdx.x;
    const size_t stride = (size_t)gridDim.x * blockDim.x;
    for (size_t i = tid; i < 16384u; i += stride) W1h[i] = (__bf16)W1[i];
    for (size_t i = tid; i < 819200u; i += stride) {            // W2T[n][k] = W2[k][n]
        size_t n = i >> 7, k = i & 127;
        W2Th[i] = (__bf16)W2[k * 6400 + n];
    }
    for (size_t i = tid; i < (size_t)E_EDGES * 128; i += stride) EAh[i] = (__bf16)EA[i];
    for (size_t i = tid; i < (size_t)N_NODES * 112; i += stride) sums[i] = 0.f;
    for (size_t i = tid; i < (size_t)N_NODES; i += stride) cnt[i] = 0.f;
}

// ---------------- kernel A: HT[tile][k][e] = relu(EA @ W1 + b1)^T (bf16) ----
__global__ __launch_bounds__(256) void mlp_h_kernel(const __bf16* __restrict__ EAh,
                                                    const __bf16* __restrict__ W1h,
                                                    const float* __restrict__ b1,
                                                    __bf16* __restrict__ HT) {
    const int lane = threadIdx.x & 31;
    const int wave = threadIdx.x >> 5;
    const int t = blockIdx.x * 8 + wave;          // 16-edge tile
    const int m = lane & 15;
    const int hs = lane >> 4;

    v16bf afr[4];                                  // edge rows (A-matrix)
#pragma unroll
    for (int kk = 0; kk < 4; ++kk) {
        const __bf16* p = EAh + ((size_t)(t * 16 + m) * 128 + kk * 32 + hs * 8);
        afr[kk] = load2x8(p, p + 16);
    }
#pragma unroll
    for (int cc = 0; cc < 8; ++cc) {
        v8f acc = {0.f,0.f,0.f,0.f,0.f,0.f,0.f,0.f};
#pragma unroll
        for (int kk = 0; kk < 4; ++kk) {
            v16bf b;
#pragma unroll
            for (int j2 = 0; j2 < 16; ++j2)
                b[j2] = W1h[(size_t)(kk * 32 + hs * 8 + (j2 & 7) + (j2 >> 3) * 16) * 128 + cc * 16 + m];
            acc = WMMA_BF16(false, afr[kk], false, b, (short)0, acc, false, false);
        }
        const float bias = b1[cc * 16 + m];
        float r[8];
#pragma unroll
        for (int j = 0; j < 8; ++j) { float x = acc[j] + bias; r[j] = x > 0.f ? x : 0.f; }
        uint4 pk;
        pk.x = packbf2(r[0], r[1]); pk.y = packbf2(r[2], r[3]);
        pk.z = packbf2(r[4], r[5]); pk.w = packbf2(r[6], r[7]);
        *(uint4*)(HT + (size_t)t * 2048 + (cc * 16 + m) * 16 + hs * 8) = pk;
    }
}

// ---------------- kernel B: fused w = h@W2+b2 + tensor-product + scatter ----
__global__ __launch_bounds__(128) void tp_conv_kernel(const __bf16* __restrict__ HT,
                                                      const unsigned char* __restrict__ W2Tb,
                                                      const float* __restrict__ b2,
                                                      const float* __restrict__ node_attr,
                                                      const int* __restrict__ edge_index,
                                                      const float* __restrict__ edge_sh,
                                                      float* __restrict__ sums,
                                                      float* __restrict__ cnt) {
    __shared__ float xs[64 * 112];                       // node_attr[dst] per block edge
    __shared__ __align__(16) unsigned char w2buf[2][16384];
    __shared__ float b2buf[2][64];
    __shared__ int dstIds[64];

    const int tid = threadIdx.x;
    const int lane = tid & 31;
    const int wave = tid >> 5;
    const int m = lane & 15;
    const int hs = lane >> 4;
    const int blockEdge0 = blockIdx.x * 64;
    const int t = blockIdx.x * 4 + wave;

    if (tid < 64) dstIds[tid] = edge_index[E_EDGES + blockEdge0 + tid];
    __syncthreads();

    { // gather node features of dst nodes into LDS (2 threads / edge, 16B pieces)
        const int el = tid >> 1, part = tid & 1;
        const float* sp = node_attr + (size_t)dstIds[el] * 112 + part * 56;
        float* dp = xs + el * 112 + part * 56;
#pragma unroll
        for (int i = 0; i < 14; ++i) {
#if HAVE_ASYNC_LDS
            async_cp16(sp + i * 4, dp + i * 4);
#else
            *(float4*)(dp + i * 4) = *(const float4*)(sp + i * 4);
#endif
        }
    }
    { // first W2T stage -> buffer 0 (16 KB, 4 chunks)
#pragma unroll
        for (int i = 0; i < 8; ++i) {
#if HAVE_ASYNC_LDS
            async_cp16(W2Tb + (size_t)(i * 128 + tid) * 16, w2buf[0] + (i * 128 + tid) * 16);
#else
            *(uint4*)(w2buf[0] + (i * 128 + tid) * 16) =
                *(const uint4*)(W2Tb + (size_t)(i * 128 + tid) * 16);
#endif
        }
        if (tid < 64) b2buf[0][tid] = b2[tid];
    }

    v16bf hfr[4];                                        // h^T fragments (B-matrix)
#pragma unroll
    for (int kk = 0; kk < 4; ++kk)
#pragma unroll
        for (int j2 = 0; j2 < 16; ++j2)
            hfr[kk][j2] = HT[(size_t)t * 2048 + (kk * 32 + hs * 8 + (j2 & 7) + (j2 >> 3) * 16) * 16 + m];

    const int ge = t * 16 + m;
    const float sh0 = edge_sh[ge * 4 + 0];
    const float s1x = edge_sh[ge * 4 + 1];
    const float s1y = edge_sh[ge * 4 + 2];
    const float s1z = edge_sh[ge * 4 + 3];
    const float N0 = 0.1118033988749895f;                // sqrt(1/80)
    const float N1 = 0.19364916731037085f;               // sqrt(3/80)
    const float IS3 = 0.5773502691896258f;
    const float cA = N0 * sh0, cD = N0 * IS3, cB = N1 * IS3, cC = N1 * IS3 * sh0;
    const int el112 = (wave * 16 + m) * 112;

    float acc0[4][8], accB[8], accC[3][8];
#pragma unroll
    for (int q = 0; q < 4; ++q)
#pragma unroll
        for (int j = 0; j < 8; ++j) acc0[q][j] = 0.f;
#pragma unroll
    for (int j = 0; j < 8; ++j) accB[j] = 0.f;
#pragma unroll
    for (int i = 0; i < 3; ++i)
#pragma unroll
        for (int j = 0; j < 8; ++j) accC[i][j] = 0.f;

#if HAVE_ASYNC_LDS
    wait_async0();
#endif
    __syncthreads();

    for (int s = 0; s < 100; ++s) {                      // 100 stages x 4 chunks
        const int cur = s & 1;
        const bool more = (s + 1) < 100;
        float b2n = 0.f;
#if !HAVE_ASYNC_LDS
        uint4 nxt[8];
#endif
        if (more) {
            const unsigned char* g = W2Tb + (size_t)(s + 1) * 16384;
            __builtin_prefetch(W2Tb + (size_t)(s + 2) * 16384, 0, 1);
#pragma unroll
            for (int i = 0; i < 8; ++i) {
#if HAVE_ASYNC_LDS
                // buf[cur^1] was last read in stage s-1 (barrier passed) -> safe
                async_cp16(g + (i * 128 + tid) * 16, w2buf[cur ^ 1] + (i * 128 + tid) * 16);
#else
                nxt[i] = *(const uint4*)(g + (i * 128 + tid) * 16);
#endif
            }
            if (tid < 64) b2n = b2[(s + 1) * 64 + tid];
        }
        const __bf16* lb = (const __bf16*)w2buf[cur];
#pragma unroll
        for (int lc = 0; lc < 4; ++lc) {
            const int c = s * 4 + lc;                    // global column-chunk id
            v8f d = {0.f,0.f,0.f,0.f,0.f,0.f,0.f,0.f};
#pragma unroll
            for (int kk = 0; kk < 4; ++kk) {             // K = 128 in 4 steps of 32
                const __bf16* p = lb + ((lc * 16 + m) * 128 + kk * 32 + hs * 8);
                const v16bf a = load2x8(p, p + 16);      // W2^T rows (A-matrix)
                d = WMMA_BF16(false, a, false, hfr[kk], (short)0, d, false, false);
            }
            float w8[8];
#pragma unroll
            for (int j = 0; j < 8; ++j) w8[j] = d[j] + b2buf[cur][lc * 16 + j + hs * 8];

            if (c < 256) {                               // wA: out0 += wA^T s * sh0
                const float f = xs[el112 + (c >> 2)] * cA;
#pragma unroll
                for (int j = 0; j < 8; ++j) acc0[lc][j] += w8[j] * f;
            } else if (c < 320) {                        // wB: sB += wB^T s
                const float f = xs[el112 + (c - 256)] * cB;
#pragma unroll
                for (int j = 0; j < 8; ++j) accB[j] += w8[j] * f;
            } else if (c < 336) {                        // wC: vC += wC^T v * sh0
                const int u = c - 320;
                const float f0 = xs[el112 + 64 + u * 3 + 0] * cC;
                const float f1 = xs[el112 + 64 + u * 3 + 1] * cC;
                const float f2 = xs[el112 + 64 + u * 3 + 2] * cC;
#pragma unroll
                for (int j = 0; j < 8; ++j) {
                    accC[0][j] += w8[j] * f0; accC[1][j] += w8[j] * f1; accC[2][j] += w8[j] * f2;
                }
            } else {                                     // wD: out0 += wD^T (v.sh1)
                const int u = (c - 336) >> 2;
                const float v0 = xs[el112 + 64 + u * 3 + 0];
                const float v1 = xs[el112 + 64 + u * 3 + 1];
                const float v2 = xs[el112 + 64 + u * 3 + 2];
                const float f = (v0 * s1x + v1 * s1y + v2 * s1z) * cD;
#pragma unroll
                for (int j = 0; j < 8; ++j) acc0[lc][j] += w8[j] * f;
            }
        }
        if (more) {
#if !HAVE_ASYNC_LDS
#pragma unroll
            for (int i = 0; i < 8; ++i)
                *(uint4*)(w2buf[cur ^ 1] + (i * 128 + tid) * 16) = nxt[i];
#endif
            if (tid < 64) b2buf[cur ^ 1][tid] = b2n;
        }
#if HAVE_ASYNC_LDS
        wait_async0();
#endif
        __syncthreads();
    }

    // segment scatter-add (each lane owns half the feature vector of its edge)
    const int srcN = edge_index[ge];
    float* sb = sums + (size_t)srcN * 112;
#pragma unroll
    for (int q = 0; q < 4; ++q)
#pragma unroll
        for (int j = 0; j < 8; ++j)
            atomicAdd(&sb[q * 16 + j + hs * 8], acc0[q][j]);
#pragma unroll
    for (int j = 0; j < 8; ++j) {
        const int w = j + hs * 8;
        atomicAdd(&sb[64 + 3 * w + 0], accB[j] * s1x + accC[0][j]);
        atomicAdd(&sb[64 + 3 * w + 1], accB[j] * s1y + accC[1][j]);
        atomicAdd(&sb[64 + 3 * w + 2], accB[j] * s1z + accC[2][j]);
    }
    if (hs == 0) atomicAdd(&cnt[srcN], 1.0f);
}

// ---------------- finalize: mean + residual --------------------------------
__global__ void finalize_kernel(const float* __restrict__ sums, const float* __restrict__ cnt,
                                const float* __restrict__ node_attr, float* __restrict__ out) {
    const size_t i = (size_t)blockIdx.x * blockDim.x + threadIdx.x;
    if (i >= (size_t)N_NODES * 112) return;
    const size_t n = i / 112;
    out[i] = sums[i] / fmaxf(cnt[n], 1.0f) + node_attr[i];
}

extern "C" void kernel_launch(void* const* d_in, const int* in_sizes, int n_in,
                              void* d_out, int out_size, void* d_ws, size_t ws_size,
                              hipStream_t stream) {
    const float* node_attr = (const float*)d_in[0];
    const int*   edge_index = (const int*)d_in[1];
    const float* edge_attr = (const float*)d_in[2];
    const float* edge_sh  = (const float*)d_in[3];
    const float* W1 = (const float*)d_in[6];
    const float* b1 = (const float*)d_in[7];
    const float* W2 = (const float*)d_in[8];
    const float* b2 = (const float*)d_in[9];

    unsigned char* ws = (unsigned char*)d_ws;
    __bf16* W1h  = (__bf16*)(ws);                                  //    32768 B
    __bf16* EAh  = (__bf16*)(ws + 32768);                          //  8388608 B
    __bf16* W2Th = (__bf16*)(ws + 32768 + 8388608);                //  1638400 B
    __bf16* HT   = (__bf16*)(ws + 32768 + 8388608 + 1638400);      //  8388608 B
    float*  sums = (float*)(ws + 32768 + 8388608 + 1638400 + 8388608);          // 1835008 B
    float*  cnt  = (float*)(ws + 32768 + 8388608 + 1638400 + 8388608 + 1835008); //  16384 B

    prep_kernel<<<2048, 256, 0, stream>>>(W1, W2, edge_attr, W1h, W2Th, EAh, sums, cnt);
    mlp_h_kernel<<<256, 256, 0, stream>>>(EAh, W1h, b1, HT);
    tp_conv_kernel<<<512, 128, 0, stream>>>(HT, (const unsigned char*)W2Th, b2, node_attr,
                                            edge_index, edge_sh, sums, cnt);
    finalize_kernel<<<(N_NODES * 112 + 255) / 256, 256, 0, stream>>>(sums, cnt, node_attr,
                                                                     (float*)d_out);
}
